// Attention_15023795602118
// MI455X (gfx1250) — matrix operand order repaired and checked by reading, not compiled, not run on hardware
//
#include <hip/hip_runtime.h>
#include <hip/hip_bf16.h>

typedef __bf16 bf16_t;
typedef __attribute__((ext_vector_type(16))) __bf16 v16bf;
typedef __attribute__((ext_vector_type(8)))  __bf16 v8bf;
typedef __attribute__((ext_vector_type(8)))  float  v8f;

#define B_  4
#define T_  2048
#define C_  2048
#define H_  16
#define D_  128
#define M_  (B_*T_)

#define LDSP 40   // LDS tile pitch in bf16 elements (80B): conflict-free, 16B aligned chunks

// ---------- bf16 helpers ----------
__device__ __forceinline__ bf16_t f2bf(float f) {
  unsigned u = __builtin_bit_cast(unsigned, f);
  unsigned r = (u + 0x7FFFu + ((u >> 16) & 1u)) >> 16;   // RNE truncation
  unsigned short h = (unsigned short)r;
  return __builtin_bit_cast(bf16_t, h);
}
__device__ __forceinline__ float bf2f(bf16_t b) {
  unsigned short h = __builtin_bit_cast(unsigned short, b);
  unsigned u = ((unsigned)h) << 16;
  return __builtin_bit_cast(float, u);
}

// ---------- WMMA wrapper ----------
__device__ __forceinline__ v8f wmma_bf16(v16bf a, v16bf b, v8f c) {
  return __builtin_amdgcn_wmma_f32_16x16x32_bf16(
      /*neg_a=*/false, a, /*neg_b=*/false, b,
      /*c_mod=*/(short)0, c, /*reuse_a=*/false, /*reuse_b=*/false);
}

// A fragment: 16x32 bf16 tile, row-major, leading dim lda.
// lanes 0-15 -> M=lane, K {0..15}; lanes 16-31 -> K {16..31}
__device__ __forceinline__ v16bf load_a16x32(const bf16_t* a, int lda, int lane) {
  int m  = lane & 15;
  int kb = (lane >> 4) << 4;          // 0 or 16
  const bf16_t* p = a + (size_t)m * lda + kb;
  v8bf lo = *(const v8bf*)(p);
  v8bf hi = *(const v8bf*)(p + 8);
  v16bf f;
#pragma unroll
  for (int i = 0; i < 8; ++i) { f[i] = lo[i]; f[i + 8] = hi[i]; }
  return f;
}

// B fragment for NT: B(k,n) = W[n][k], W row-major (N x K), contiguous in k.
__device__ __forceinline__ v16bf load_b32x16_nt(const bf16_t* w, int ldw, int lane) {
  int n  = lane & 15;
  int kb = (lane >> 4) << 4;          // 0 or 16
  return *(const v16bf*)(w + (size_t)n * ldw + kb);
}

__device__ __forceinline__ void cvt_store(float* p, float v)  { *p = v; }
__device__ __forceinline__ void cvt_store(bf16_t* p, float v) { *p = f2bf(v); }

// ---------- CDNA5 async global->LDS copy (ASYNCcnt-tracked) ----------
__device__ __forceinline__ void async_b128(unsigned lds_byte, const bf16_t* g) {
  unsigned long long ga = (unsigned long long)g;
  asm volatile("global_load_async_to_lds_b128 %0, %1, off"
               :
               : "v"(lds_byte), "v"(ga)
               : "memory");
}
__device__ __forceinline__ void wait_async0() {
  asm volatile("s_wait_asynccnt 0x0" ::: "memory");
}

// Copy a 128-row x 32-col bf16 tile (row stride ldg) into LDS (pitch LDSP elems).
// 256 threads x 2 iters x 16B = 8KB.
__device__ __forceinline__ void async_copy_tile128x32(const bf16_t* gsrc, int ldg,
                                                      unsigned lds_base, int tid) {
#pragma unroll
  for (int it = 0; it < 2; ++it) {
    int slot = tid + it * 256;
    int row  = slot >> 2;          // 0..127
    int ch   = slot & 3;           // 16B chunk (8 bf16)
    unsigned lds = lds_base + (unsigned)(row * (LDSP * 2) + ch * 16);
    async_b128(lds, gsrc + (size_t)row * ldg + ch * 8);
  }
}

// ---------- cast kernel ----------
__global__ __launch_bounds__(256) void cast_f32_bf16(const float* __restrict__ src,
                                                     bf16_t* __restrict__ dst, int n) {
  int i = blockIdx.x * blockDim.x + threadIdx.x;
  if (i < n) dst[i] = f2bf(src[i]);
}

// ---------- RoPE (in-place on bf16 Q and K, layout (B,T,H,D)) ----------
__global__ __launch_bounds__(256) void rope_kernel(bf16_t* __restrict__ q,
                                                   bf16_t* __restrict__ k, int n) {
  int i = blockIdx.x * blockDim.x + threadIdx.x;
  if (i >= n) return;
  int d = i & 63;
  int h = (i >> 6) & 15;
  int t = (i >> 10) & 2047;
  int b = i >> 21;
  float freq = __expf(-(float)d * (9.210340371976184f / 64.f)); // 10000^(-d/64)
  float ang = (float)t * freq;
  float s, c;
  __sincosf(ang, &s, &c);
  size_t base = (((size_t)b * T_ + t) * C_) + (size_t)h * D_ + d;
  float x1 = bf2f(q[base]), x2 = bf2f(q[base + 64]);
  q[base]      = f2bf(x1 * c - x2 * s);
  q[base + 64] = f2bf(x1 * s + x2 * c);
  float y1 = bf2f(k[base]), y2 = bf2f(k[base + 64]);
  k[base]      = f2bf(y1 * c - y2 * s);
  k[base + 64] = f2bf(y1 * s + y2 * c);
}

// ---------- NT GEMM: C[M,N] = A[M,K] * W[N,K]^T ----------
// block = 256 threads (8 waves); block tile 128(M) x 128(N); wave tile 32x64.
// Double-buffered LDS staging via global_load_async_to_lds_b128.
template <typename OutT>
__global__ __launch_bounds__(256) void gemm_nt_wmma(const bf16_t* __restrict__ A,
                                                    const bf16_t* __restrict__ W,
                                                    OutT* __restrict__ Cmat,
                                                    int M, int N, int K) {
  __shared__ __align__(128) bf16_t shA[2][128 * LDSP];   // 10KB each
  __shared__ __align__(128) bf16_t shB[2][128 * LDSP];

  const int tid  = threadIdx.x;
  const int lane = tid & 31;
  const int wv   = tid >> 5;
  const int wm   = wv & 3;           // 4 M slots of 32
  const int wn   = wv >> 2;          // 2 N slots of 64
  const int rowB = blockIdx.x * 128;
  const int colB = blockIdx.y * 128;

  const bf16_t* ablk = A + (size_t)rowB * K;
  const bf16_t* bblk = W + (size_t)colB * K;

  unsigned baseA[2], baseB[2];
  baseA[0] = (unsigned)(unsigned long long)&shA[0][0];
  baseA[1] = (unsigned)(unsigned long long)&shA[1][0];
  baseB[0] = (unsigned)(unsigned long long)&shB[0][0];
  baseB[1] = (unsigned)(unsigned long long)&shB[1][0];

  v8f zero = {};
  v8f acc[2][4];
#pragma unroll
  for (int mt = 0; mt < 2; ++mt)
#pragma unroll
    for (int nt = 0; nt < 4; ++nt) acc[mt][nt] = zero;

  // prologue: stage k=0 tiles
  async_copy_tile128x32(ablk, K, baseA[0], tid);
  async_copy_tile128x32(bblk, K, baseB[0], tid);
  wait_async0();
  __syncthreads();

  for (int k = 0; k < K; k += 32) {
    const int cur = (k >> 5) & 1;
    if (k + 32 < K) {
      async_copy_tile128x32(ablk + k + 32, K, baseA[cur ^ 1], tid);
      async_copy_tile128x32(bblk + k + 32, K, baseB[cur ^ 1], tid);
    }

    const bf16_t* sA = &shA[cur][0];
    const bf16_t* sB = &shB[cur][0];
    v16bf af[2], bfr[4];
#pragma unroll
    for (int mt = 0; mt < 2; ++mt)
      af[mt] = load_a16x32(sA + (size_t)(wm * 32 + mt * 16) * LDSP, LDSP, lane);
#pragma unroll
    for (int nt = 0; nt < 4; ++nt)
      bfr[nt] = load_b32x16_nt(sB + (size_t)(wn * 64 + nt * 16) * LDSP, LDSP, lane);
#pragma unroll
    for (int mt = 0; mt < 2; ++mt)
#pragma unroll
      for (int nt = 0; nt < 4; ++nt)
        acc[mt][nt] = wmma_bf16(af[mt], bfr[nt], acc[mt][nt]);

    wait_async0();          // next buffer resident
    __syncthreads();        // everyone done reading cur before it is overwritten
  }

  const int n  = lane & 15;
  const int m0 = (lane >> 4) << 3;
  const int row0 = rowB + wm * 32;
  const int col0 = colB + wn * 64;
#pragma unroll
  for (int mt = 0; mt < 2; ++mt)
#pragma unroll
    for (int nt = 0; nt < 4; ++nt)
#pragma unroll
      for (int r = 0; r < 8; ++r)
        cvt_store(&Cmat[(size_t)(row0 + mt * 16 + m0 + r) * N + col0 + nt * 16 + n],
                  acc[mt][nt][r]);
}

// ---------- causal flash attention ----------
// grid = (T/64, B*H), block = 128 threads (4 waves). Wave wv handles 16 q rows.
__global__ __launch_bounds__(128) void attn_wmma(const bf16_t* __restrict__ Q,
                                                 const bf16_t* __restrict__ Kt,
                                                 const bf16_t* __restrict__ V,
                                                 bf16_t* __restrict__ O) {
  __shared__ __align__(64) bf16_t vsh[D_ * 64];     // V^T tile: [d][key], 16KB
  __shared__ __align__(64) bf16_t psh[4][16 * 64];  // per-wave P tile, 8KB

  const int lane = threadIdx.x & 31;
  const int wv   = threadIdx.x >> 5;
  const int q0   = blockIdx.x * 64;
  const int b    = blockIdx.y >> 4;
  const int h    = blockIdx.y & 15;
  const float scale = 0.08838834764831845f;         // 1/sqrt(128)

  const bf16_t* qbase = Q + ((size_t)(b * T_ + q0 + wv * 16)) * C_ + (size_t)h * D_;
  const bf16_t* kbase = Kt + ((size_t)b * T_) * C_ + (size_t)h * D_;
  const bf16_t* vbase = V + ((size_t)b * T_) * C_ + (size_t)h * D_;

  v16bf qf[4];
#pragma unroll
  for (int ks = 0; ks < 4; ++ks) qf[ks] = load_a16x32(qbase + ks * 32, C_, lane);

  v8f zero = {};
  v8f o[8];
#pragma unroll
  for (int d = 0; d < 8; ++d) o[d] = zero;
  float mstat[8], lstat[8];
#pragma unroll
  for (int r = 0; r < 8; ++r) { mstat[r] = -1e30f; lstat[r] = 0.f; }

  const int nl = lane & 15;
  const int mh = (lane >> 4) << 3;

  for (int j0 = 0; j0 <= q0; j0 += 64) {
    // cooperative stage of V^T[d][key] into LDS
    for (int idx = threadIdx.x; idx < 64 * 16; idx += 128) {
      int key = idx >> 4;
      int dc  = (idx & 15) << 3;
      v8bf vvv = *(const v8bf*)(vbase + (size_t)(j0 + key) * C_ + dc);
#pragma unroll
      for (int i = 0; i < 8; ++i) vsh[(dc + i) * 64 + key] = vvv[i];
    }
    __syncthreads();

    // S = (Q K^T) * scale, 4 tiles of 16 keys each
    v8f st[4];
#pragma unroll
    for (int t = 0; t < 4; ++t) {
      v8f s = zero;
      const bf16_t* kp = kbase + (size_t)(j0 + t * 16) * C_;
#pragma unroll
      for (int ks = 0; ks < 4; ++ks)
        s = wmma_bf16(qf[ks], load_b32x16_nt(kp + ks * 32, C_, lane), s);
#pragma unroll
      for (int r = 0; r < 8; ++r) {
        int nglob = j0 + t * 16 + nl;
        int mglob = q0 + wv * 16 + mh + r;
        st[t][r] = (nglob > mglob) ? -1e30f : s[r] * scale;
      }
    }

    // online softmax (rows live across 16-lane groups of the C layout)
    float corr[8];
#pragma unroll
    for (int r = 0; r < 8; ++r) {
      float rowmax = -1e30f;
#pragma unroll
      for (int t = 0; t < 4; ++t) rowmax = fmaxf(rowmax, st[t][r]);
#pragma unroll
      for (int msk = 1; msk < 16; msk <<= 1)
        rowmax = fmaxf(rowmax, __shfl_xor(rowmax, msk, 32));
      float mnew = fmaxf(mstat[r], rowmax);
      corr[r] = __expf(mstat[r] - mnew);
      mstat[r] = mnew;
      float rsum = 0.f;
#pragma unroll
      for (int t = 0; t < 4; ++t) {
        float p = __expf(st[t][r] - mnew);
        st[t][r] = p;
        rsum += p;
      }
#pragma unroll
      for (int msk = 1; msk < 16; msk <<= 1)
        rsum += __shfl_xor(rsum, msk, 32);
      lstat[r] = lstat[r] * corr[r] + rsum;
    }
#pragma unroll
    for (int d = 0; d < 8; ++d)
#pragma unroll
      for (int r = 0; r < 8; ++r) o[d][r] *= corr[r];

    // P -> LDS (bf16), then reload as A fragments
#pragma unroll
    for (int t = 0; t < 4; ++t)
#pragma unroll
      for (int r = 0; r < 8; ++r)
        psh[wv][(mh + r) * 64 + t * 16 + nl] = f2bf(st[t][r]);

    // O += P * V  (A = P 16x64, B = V^T from LDS; contiguous ds vector loads)
#pragma unroll
    for (int ks = 0; ks < 2; ++ks) {
      v16bf pa = load_a16x32(&psh[wv][0] + ks * 32, 64, lane);
      int kb = ((lane >> 4) << 4) + ks * 32;
#pragma unroll
      for (int d = 0; d < 8; ++d) {
        v16bf vbf = *(const v16bf*)(&vsh[(d * 16 + nl) * 64 + kb]);
        o[d] = wmma_bf16(pa, vbf, o[d]);
      }
    }
    __syncthreads();
  }

  // normalize and write attn output in (B,T,H*D) layout
  bf16_t* op = O + ((size_t)(b * T_ + q0 + wv * 16)) * C_ + (size_t)h * D_;
#pragma unroll
  for (int r = 0; r < 8; ++r) {
    float inv = 1.f / lstat[r];
#pragma unroll
    for (int d = 0; d < 8; ++d)
      op[(size_t)(mh + r) * C_ + d * 16 + nl] = f2bf(o[d][r] * inv);
  }
}

// ---------- launch ----------
extern "C" void kernel_launch(void* const* d_in, const int* in_sizes, int n_in,
                              void* d_out, int out_size, void* d_ws, size_t ws_size,
                              hipStream_t stream) {
  (void)in_sizes; (void)n_in; (void)out_size; (void)ws_size;
  const float* x  = (const float*)d_in[0];
  const float* wq = (const float*)d_in[1];
  const float* wk = (const float*)d_in[2];
  const float* wv = (const float*)d_in[3];
  const float* wo = (const float*)d_in[4];
  float* out = (float*)d_out;

  char* ws = (char*)d_ws;
  const size_t SZ_X = (size_t)M_ * C_ * sizeof(bf16_t);  // 32 MB
  const size_t SZ_W = (size_t)C_ * C_ * sizeof(bf16_t);  //  8 MB
  bf16_t* xb  = (bf16_t*)(ws);
  bf16_t* wqb = (bf16_t*)(ws + SZ_X);
  bf16_t* wkb = (bf16_t*)(ws + SZ_X + 1 * SZ_W);
  bf16_t* wvb = (bf16_t*)(ws + SZ_X + 2 * SZ_W);
  bf16_t* wob = (bf16_t*)(ws + SZ_X + 3 * SZ_W);
  bf16_t* qb  = (bf16_t*)(ws + 1 * SZ_X + 4 * SZ_W);
  bf16_t* kb  = (bf16_t*)(ws + 2 * SZ_X + 4 * SZ_W);
  bf16_t* vb  = (bf16_t*)(ws + 3 * SZ_X + 4 * SZ_W);
  bf16_t* ab  = (bf16_t*)(ws + 4 * SZ_X + 4 * SZ_W);   // total 192 MB

  const int nx = M_ * C_;
  const int nw = C_ * C_;
  cast_f32_bf16<<<(nx + 255) / 256, 256, 0, stream>>>(x,  xb,  nx);
  cast_f32_bf16<<<(nw + 255) / 256, 256, 0, stream>>>(wq, wqb, nw);
  cast_f32_bf16<<<(nw + 255) / 256, 256, 0, stream>>>(wk, wkb, nw);
  cast_f32_bf16<<<(nw + 255) / 256, 256, 0, stream>>>(wv, wvb, nw);
  cast_f32_bf16<<<(nw + 255) / 256, 256, 0, stream>>>(wo, wob, nw);

  dim3 ggrid(M_ / 128, C_ / 128);
  gemm_nt_wmma<bf16_t><<<ggrid, 256, 0, stream>>>(xb, wqb, qb, M_, C_, C_);
  gemm_nt_wmma<bf16_t><<<ggrid, 256, 0, stream>>>(xb, wkb, kb, M_, C_, C_);
  gemm_nt_wmma<bf16_t><<<ggrid, 256, 0, stream>>>(xb, wvb, vb, M_, C_, C_);

  const int nr = B_ * T_ * H_ * (D_ / 2);
  rope_kernel<<<(nr + 255) / 256, 256, 0, stream>>>(qb, kb, nr);

  attn_wmma<<<dim3(T_ / 64, B_ * H_), 128, 0, stream>>>(qb, kb, vb, ab);

  gemm_nt_wmma<float><<<ggrid, 256, 0, stream>>>(ab, wob, out, M_, C_, C_);
}
